// WMSA_75625784148262
// MI455X (gfx1250) — matrix operand order repaired and checked by reading, not compiled, not run on hardware
//
#include <hip/hip_runtime.h>
#include <stdint.h>

typedef __attribute__((ext_vector_type(16))) _Float16 v16h;
typedef __attribute__((ext_vector_type(8)))  _Float16 v8h;
typedef __attribute__((ext_vector_type(8)))  float    v8f;

union Vh16 { v16h v; v8h p[2]; uint32_t u[8]; };

// ---------------- workspace layout ----------------
// [0, 221184)           : wqkvT f16  [576][192]   (B-matrix, N-major)
// [221184, 294912)      : woutT f16  [192][192]
// [294912, 393216)      : attn bias  f32 [6][64][64]
#define WS_WQKVT  0
#define WS_WOUTT  221184
#define WS_ABIAS  294912

// ---------------- prologue: convert+transpose weights to f16 ----------------
__global__ void prep_weights(const float* __restrict__ w_qkv,
                             const float* __restrict__ w_out,
                             _Float16* __restrict__ wqkvT,
                             _Float16* __restrict__ woutT) {
  int idx = blockIdx.x * 256 + threadIdx.x;          // 147456 total
  if (idx < 576 * 192) {
    int n = idx / 192, k = idx % 192;
    wqkvT[idx] = (_Float16)w_qkv[k * 576 + n];
  } else {
    int j = idx - 576 * 192;                         // 36864
    int n = j / 192, k = j % 192;
    woutT[j] = (_Float16)w_out[k * 192 + n];
  }
}

// ---------------- prologue: expand relative-position bias ----------------
__global__ void prep_bias(const float* __restrict__ rel_pos,
                          float* __restrict__ abias) {
  int idx = blockIdx.x * 256 + threadIdx.x;          // 6*64*64 = 24576
  int h = idx >> 12;
  int rem = idx & 4095;
  int p = rem >> 6, q = rem & 63;
  int r0 = (p >> 3) - (q >> 3) + 7;
  int r1 = (p & 7) - (q & 7) + 7;
  abias[idx] = rel_pos[h * 225 + r0 * 15 + r1];
}

// ---------------- helpers ----------------
__device__ __forceinline__ uint32_t pk2(float a, float b) {
  union { _Float16 h[2]; uint32_t u; } t;
  t.h[0] = (_Float16)a; t.h[1] = (_Float16)b;
  return t.u;
}
__device__ __forceinline__ uint4 pack8(v8f a, float s) {
  uint4 r;
  r.x = pk2(a[0] * s, a[1] * s);
  r.y = pk2(a[2] * s, a[3] * s);
  r.z = pk2(a[4] * s, a[5] * s);
  r.w = pk2(a[6] * s, a[7] * s);
  return r;
}
__device__ __forceinline__ uint4 xor16(uint4 a) {
  uint4 r;
  r.x = (uint32_t)__shfl_xor((int)a.x, 16, 32);
  r.y = (uint32_t)__shfl_xor((int)a.y, 16, 32);
  r.z = (uint32_t)__shfl_xor((int)a.z, 16, 32);
  r.w = (uint32_t)__shfl_xor((int)a.w, 16, 32);
  return r;
}
#define WMMA(A, B, C) __builtin_amdgcn_wmma_f32_16x16x32_f16(false, (A), false, (B), (short)0, (C), false, false)

// ---------------- fused W-MSA kernel: one block = one window ----------------
__global__ __launch_bounds__(256, 1) void wmsa_fused(
    const float* __restrict__ x, const float* __restrict__ b_qkv,
    const float* __restrict__ b_out,
    const _Float16* __restrict__ wqkvT, const _Float16* __restrict__ woutT,
    const float* __restrict__ abias, float* __restrict__ out) {
  __shared__ _Float16 xA[64 * 192];     // x window (A matrix), later attn output
  __shared__ _Float16 qkvS[64 * 576];   // qkv, row-major [token][chan]

  const int tid = threadIdx.x;
  const int wave = tid >> 5;
  const int lane = tid & 31;
  const int l16 = lane & 15;
  const int hiH = (lane >> 4) & 1;

  const int blk = blockIdx.x;
  const int batch = blk >> 10;
  const int widx = blk & 1023;
  const int wr = widx >> 5;
  const int wc = widx & 31;

  // ---- phase 0: gather rolled window, f32 -> f16 into LDS ----
  {
    const float4* xg = (const float4*)x;
#pragma unroll
    for (int i = 0; i < 12; ++i) {
      int ve = tid + i * 256;                  // 3072 float4 total
      int token = ve / 48;
      int c4 = ve % 48;
      int ty = token >> 3, tx = token & 7;
      int gy = (wr * 8 + ty + 4) & 255;        // roll(-4) folded in
      int gx = (wc * 8 + tx + 4) & 255;
      float4 v = xg[(size_t)(((batch << 8) + gy) * 256 + gx) * 48 + c4];
      union { _Float16 h[4]; uint2 u; } p;
      p.h[0] = (_Float16)v.x; p.h[1] = (_Float16)v.y;
      p.h[2] = (_Float16)v.z; p.h[3] = (_Float16)v.w;
      *(uint2*)&xA[token * 192 + c4 * 4] = p.u;
    }
  }
  __syncthreads();

  // ---- phase 1: QKV GEMM (64x192)@(192x576) -> qkvS f16 ----
  for (int nt = wave; nt < 36; nt += 8) {
    const int nIdx = nt * 16 + l16;
    const _Float16* bp = wqkvT + nIdx * 192 + (hiH ? 16 : 0);
    v16h bfrag[6];
#pragma unroll
    for (int kt = 0; kt < 6; ++kt) bfrag[kt] = *(const v16h*)(bp + kt * 32);
    const float bias = b_qkv[nIdx];
#pragma unroll
    for (int mt = 0; mt < 4; ++mt) {
      v8f acc;
#pragma unroll
      for (int r = 0; r < 8; ++r) acc[r] = bias;
      const int row = mt * 16 + l16;
#pragma unroll
      for (int kt = 0; kt < 6; ++kt) {
        const _Float16* ap = &xA[row * 192 + kt * 32 + (hiH ? 16 : 0)];
        Vh16 af; af.p[0] = *(const v8h*)ap; af.p[1] = *(const v8h*)(ap + 8);
        acc = WMMA(af.v, bfrag[kt], acc);
      }
      const int tok0 = mt * 16 + (hiH ? 8 : 0);
#pragma unroll
      for (int r = 0; r < 8; ++r) qkvS[(tok0 + r) * 576 + nIdx] = (_Float16)acc[r];
    }
  }
  __syncthreads();

  // ---- phase 2: attention, one wave per head (waves 6,7 idle) ----
  if (wave < 6) {
    const int h = wave;
    const int bq = h * 32, bk = 192 + h * 32, bv = 384 + h * 32;
    const float scale = 0.17677669529663687f;  // 1/sqrt(32)

    // K rows as A-fragments (S^T = K * Q^T)
    v16h kfrag[4];
#pragma unroll
    for (int mt = 0; mt < 4; ++mt) {
      const _Float16* ap = &qkvS[(mt * 16 + l16) * 576 + bk + (hiH ? 16 : 0)];
      Vh16 t; t.p[0] = *(const v8h*)ap; t.p[1] = *(const v8h*)(ap + 8);
      kfrag[mt] = t.v;
    }
    // V^T as A-fragments for out^T = V^T * P^T (strided LDS gather)
    v16h vfrag[2][2];
    const int lgo = hiH ? 16 : 0;
#pragma unroll
    for (int mt2 = 0; mt2 < 2; ++mt2) {
      const int dim = mt2 * 16 + l16;
#pragma unroll
      for (int kt = 0; kt < 2; ++kt) {
        Vh16 t;
#pragma unroll
        for (int s = 0; s < 8; ++s) {
          t.v[s]     = qkvS[(kt * 32 + lgo + s) * 576 + bv + dim];
          t.v[s + 8] = qkvS[(kt * 32 + lgo + 8 + s) * 576 + bv + dim];
        }
        vfrag[mt2][kt] = t.v;
      }
    }
    const bool lastR = (wr == 31), lastC = (wc == 31);

#pragma unroll
    for (int nt = 0; nt < 4; ++nt) {
      const int iTok = nt * 16 + l16;                 // query token of this lane
      v16h qfrag = *(const v16h*)&qkvS[iTok * 576 + bq + (hiH ? 16 : 0)];
      v8f stc[4];
#pragma unroll
      for (int mt = 0; mt < 4; ++mt) {
        v8f z;
#pragma unroll
        for (int r = 0; r < 8; ++r) z[r] = 0.0f;
        stc[mt] = WMMA(kfrag[mt], qfrag, z);          // S^T tile: rows j, col i
      }
      // scale + rel-pos bias + shift mask
      const bool iR = ((iTok >> 3) < 4), iC = ((iTok & 7) < 4);
#pragma unroll
      for (int mt = 0; mt < 4; ++mt) {
        const int jb = mt * 16 + (hiH ? 8 : 0);
        const float* bsrc = abias + ((h * 64 + iTok) * 64 + jb);
        const float4 b0 = *(const float4*)bsrc;
        const float4 b1 = *(const float4*)(bsrc + 4);
        float bb[8] = {b0.x, b0.y, b0.z, b0.w, b1.x, b1.y, b1.z, b1.w};
#pragma unroll
        for (int r = 0; r < 8; ++r) {
          const int j = jb + r;
          float v = stc[mt][r] * scale + bb[r];
          if ((lastR && (((j >> 3) < 4) != iR)) ||
              (lastC && (((j & 7) < 4) != iC))) v = -1e30f;
          stc[mt][r] = v;
        }
      }
      // softmax over j: in-lane (32 vals) + one partner-lane exchange
      float mx = -3.0e38f;
#pragma unroll
      for (int mt = 0; mt < 4; ++mt)
#pragma unroll
        for (int r = 0; r < 8; ++r) mx = fmaxf(mx, stc[mt][r]);
      mx = fmaxf(mx, __shfl_xor(mx, 16, 32));
      float sum = 0.0f;
#pragma unroll
      for (int mt = 0; mt < 4; ++mt)
#pragma unroll
        for (int r = 0; r < 8; ++r) {
          float e = __expf(stc[mt][r] - mx);
          stc[mt][r] = e; sum += e;
        }
      sum += __shfl_xor(sum, 16, 32);
      const float inv = 1.0f / sum;

      // P^T accumulator -> B-fragments via lane shuffles; out^T = V^T * P^T
      v8f o[2];
#pragma unroll
      for (int m = 0; m < 2; ++m)
#pragma unroll
        for (int r = 0; r < 8; ++r) o[m][r] = 0.0f;
#pragma unroll
      for (int kt = 0; kt < 2; ++kt) {
        uint4 pa = pack8(stc[2 * kt], inv);
        uint4 pb = pack8(stc[2 * kt + 1], inv);
        uint4 pax = xor16(pa);
        uint4 pbx = xor16(pb);
        uint4 lo = hiH ? pbx : pa;
        uint4 hi = hiH ? pb : pax;
        Vh16 bf;
        bf.u[0] = lo.x; bf.u[1] = lo.y; bf.u[2] = lo.z; bf.u[3] = lo.w;
        bf.u[4] = hi.x; bf.u[5] = hi.y; bf.u[6] = hi.z; bf.u[7] = hi.w;
#pragma unroll
        for (int mt2 = 0; mt2 < 2; ++mt2)
          o[mt2] = WMMA(vfrag[mt2][kt], bf.v, o[mt2]);
      }
      // out^T tile -> attn output in xA[token][chan] (packed b128 stores)
#pragma unroll
      for (int mt2 = 0; mt2 < 2; ++mt2) {
        uint4 pk = pack8(o[mt2], 1.0f);
        *(uint4*)&xA[iTok * 192 + h * 32 + mt2 * 16 + (hiH ? 8 : 0)] = pk;
      }
    }
  }
  __syncthreads();

  // ---- phase 3: output projection + un-roll scatter ----
  for (int t = wave; t < 48; t += 8) {
    const int mt = t / 12;
    const int nt = t % 12;
    const int nIdx = nt * 16 + l16;
    const _Float16* bp = woutT + nIdx * 192 + (hiH ? 16 : 0);
    v16h bfrag[6];
#pragma unroll
    for (int kt = 0; kt < 6; ++kt) bfrag[kt] = *(const v16h*)(bp + kt * 32);
    v8f acc;
    const float bias = b_out[nIdx];
#pragma unroll
    for (int r = 0; r < 8; ++r) acc[r] = bias;
    const int row = mt * 16 + l16;
#pragma unroll
    for (int kt = 0; kt < 6; ++kt) {
      const _Float16* ap = &xA[row * 192 + kt * 32 + (hiH ? 16 : 0)];
      Vh16 af; af.p[0] = *(const v8h*)ap; af.p[1] = *(const v8h*)(ap + 8);
      acc = WMMA(af.v, bfrag[kt], acc);
    }
    const int tok0 = mt * 16 + (hiH ? 8 : 0);
#pragma unroll
    for (int r = 0; r < 8; ++r) {
      const int p = tok0 + r;
      const int ty = p >> 3, tx = p & 7;
      const int gy = (wr * 8 + ty + 4) & 255;  // roll(+4) folded in
      const int gx = (wc * 8 + tx + 4) & 255;
      out[(size_t)(((batch << 8) + gy) * 256 + gx) * 192 + nIdx] = acc[r];
    }
  }
}

// ---------------- launcher ----------------
extern "C" void kernel_launch(void* const* d_in, const int* in_sizes, int n_in,
                              void* d_out, int out_size, void* d_ws, size_t ws_size,
                              hipStream_t stream) {
  const float* x       = (const float*)d_in[0];
  const float* w_qkv   = (const float*)d_in[1];
  const float* b_qkv   = (const float*)d_in[2];
  const float* rel_pos = (const float*)d_in[3];
  const float* w_out   = (const float*)d_in[4];
  const float* b_out   = (const float*)d_in[5];
  float* out = (float*)d_out;
  char* ws = (char*)d_ws;
  _Float16* wqkvT = (_Float16*)(ws + WS_WQKVT);
  _Float16* woutT = (_Float16*)(ws + WS_WOUTT);
  float* abias    = (float*)(ws + WS_ABIAS);

  prep_weights<<<576, 256, 0, stream>>>(w_qkv, w_out, wqkvT, woutT);
  prep_bias<<<96, 256, 0, stream>>>(rel_pos, abias);
  wmsa_fused<<<4096, 256, 0, stream>>>(x, b_qkv, b_out, wqkvT, woutT, abias, out);
}
